// GCN_6493990551891
// MI455X (gfx1250) — matrix operand order repaired and checked by reading, not compiled, not run on hardware
//
#include <hip/hip_runtime.h>

typedef _Float16 half_t;
typedef __attribute__((ext_vector_type(16))) _Float16 v16h;
typedef __attribute__((ext_vector_type(8)))  _Float16 v8h;
typedef __attribute__((ext_vector_type(8)))  float    v8f;
typedef __attribute__((ext_vector_type(4)))  float    v4f;

#define B_GRAPHS 16384
#define N_NODES  16
#define D_IN     256
#define D_H      128
#define D_H2     64
#define D_OUT    32
#define WAVES    8

// workspace layout (in halves):
//   wt1[n][k] : [128][256] at offset 0      (W1 transposed, f16)
//   wt2[n][k] : [ 64][128] at offset 32768
//   wt3[n][k] : [ 32][ 64] at offset 40960
#define WT2_OFF 32768
#define WT3_OFF 40960

__global__ __launch_bounds__(256)
void gcn_prep(const float* __restrict__ W1, const float* __restrict__ W2,
              const float* __restrict__ W3, half_t* __restrict__ ws) {
  int i = blockIdx.x * blockDim.x + threadIdx.x;
  if (i < D_H * D_IN) {              // wt1[n][k] = W1[k][n]
    int n = i / D_IN, k = i % D_IN;
    ws[i] = (half_t)W1[k * D_H + n];
  }
  if (i < D_H2 * D_H) {              // wt2[n][k] = W2[k][n]
    int n = i / D_H, k = i % D_H;
    ws[WT2_OFF + i] = (half_t)W2[k * D_H2 + n];
  }
  if (i < D_OUT * D_H2) {            // wt3[n][k] = W3[k][n]
    int n = i / D_H2, k = i % D_H2;
    ws[WT3_OFF + i] = (half_t)W3[k * D_OUT + n];
  }
}

__device__ __forceinline__ v8f wmma_f16(v16h a, v16h b, v8f c) {
  // D = A(16x32 f16) * B(32x16 f16) + C(16x16 f32)
  return __builtin_amdgcn_wmma_f32_16x16x32_f16(
      /*neg_a=*/false, a, /*neg_b=*/false, b,
      /*c_mod=*/(short)0, c, /*reuse_a=*/false, /*reuse_b=*/false);
}

// A-fragment (16x32, f16) from row-major f32. Per ISA layout:
//   lanes 0-15 : halves 0-7 = K k0+0..7 , halves 8-15 = K k0+16..23
//   lanes16-31 : halves 0-7 = K k0+8..15, halves 8-15 = K k0+24..31
// caller passes p = rowptr + k0 + khalf*8
__device__ __forceinline__ v16h a_frag_f32(const float* p) {
  v4f f0 = *(const v4f*)(p);
  v4f f1 = *(const v4f*)(p + 4);
  v4f f2 = *(const v4f*)(p + 16);
  v4f f3 = *(const v4f*)(p + 20);
  v16h a;
#pragma unroll
  for (int i = 0; i < 4; ++i) {
    a[i]      = (half_t)f0[i];
    a[4 + i]  = (half_t)f1[i];
    a[8 + i]  = (half_t)f2[i];
    a[12 + i] = (half_t)f3[i];
  }
  return a;
}

// A-fragment (16 halves) from row-major f16 in LDS (same layout as above)
__device__ __forceinline__ v16h a_frag_h(const half_t* p) {
  v8h lo = *(const v8h*)(p);
  v8h hi = *(const v8h*)(p + 16);
  return __builtin_shufflevector(lo, hi, 0, 1, 2, 3, 4, 5, 6, 7,
                                 8, 9, 10, 11, 12, 13, 14, 15);
}

__global__ __launch_bounds__(256)
void gcn_main(const float* __restrict__ X, const float* __restrict__ adj,
              const float* __restrict__ b1, const float* __restrict__ b2,
              const float* __restrict__ b3, const half_t* __restrict__ ws,
              float* __restrict__ out) {
  // per-wave LDS: H staging (16x128 f16) + 16x16 tile staging (col-major f16)
  __shared__ __attribute__((aligned(32))) half_t lds_h[WAVES][N_NODES * D_H];
  __shared__ __attribute__((aligned(32))) half_t lds_s[WAVES][N_NODES * N_NODES];

  const int lane  = threadIdx.x & 31;
  const int wave  = threadIdx.x >> 5;
  const int l15   = lane & 15;
  const int khalf = lane >> 4;
  const int g     = blockIdx.x * WAVES + wave;

  const half_t* wt1 = ws;
  const half_t* wt2 = ws + WT2_OFF;
  const half_t* wt3 = ws + WT3_OFF;

  const float* Xg = X + (size_t)g * (N_NODES * D_IN);
  const float* Ag = adj + (size_t)g * (N_NODES * N_NODES);
  half_t* hb = lds_h[wave];
  half_t* st = lds_s[wave];

  const v8f  vzero = {};
  const v16h hzero = {};

  // ---- adj as A-fragment, K padded 16 -> 32 with zeros ----
  v16h afrag = {};
  {
    const float* p = Ag + l15 * 16 + khalf * 8;   // lanes>=16 grab K=8..15
    v4f f0 = *(const v4f*)(p);
    v4f f1 = *(const v4f*)(p + 4);
#pragma unroll
    for (int i = 0; i < 4; ++i) {
      afrag[i]     = (half_t)f0[i];
      afrag[4 + i] = (half_t)f1[i];
    }
    // halves 8..15 (K=16..31) stay zero
  }

  // ---- X (16x256) as 8 A-fragments, f32 -> f16 (each element loaded once) ----
  v16h ax[8];
#pragma unroll
  for (int kc = 0; kc < 8; ++kc)
    ax[kc] = a_frag_f32(Xg + l15 * D_IN + kc * 32 + khalf * 8);

  // ================= Layer 1: relu(adj @ (X@W1) + b1) =================
#pragma unroll
  for (int nt = 0; nt < 8; ++nt) {
    v8f acc = vzero;
#pragma unroll
    for (int kc = 0; kc < 8; ++kc) {
      // B operand: wt1 row n holds column n of W1 (two 16B chunks per lane, K k0+8h.. and k0+16+8h..)
      const half_t* bp = wt1 + (nt * 16 + l15) * D_IN + kc * 32 + khalf * 8;
      v16h b = __builtin_shufflevector(*(const v8h*)(bp), *(const v8h*)(bp + 16), 0, 1, 2, 3, 4, 5, 6, 7, 8, 9, 10, 11, 12, 13, 14, 15);
      acc = wmma_f16(ax[kc], b, acc);
    }
    // stage S1 tile -> LDS column-major f16 (one b128 store per lane)
    {
      v8h h;
#pragma unroll
      for (int i = 0; i < 8; ++i) h[i] = (half_t)acc[i];
      *(v8h*)(st + l15 * 16 + khalf * 8) = h;   // col n=l15, rows m0..m0+7
    }
    // re-read as B operand (halves 0-7 = K 8*khalf..+7; K 16..31 zero)
    v16h bs = hzero;
    bs.lo = *(const v8h*)(st + l15 * 16 + khalf * 8);
    v8f h1 = wmma_f16(afrag, bs, vzero);
    // bias + relu -> H1 row-major f16 in LDS
    {
      int n = nt * 16 + l15;
      float bv = b1[n];
      int m0 = khalf * 8;
#pragma unroll
      for (int r = 0; r < 8; ++r) {
        float v = h1[r] + bv;
        hb[(m0 + r) * D_H + n] = (half_t)(v > 0.f ? v : 0.f);
      }
    }
  }

  // ================= Layer 2: relu(adj @ (H1@W2) + b2) =================
  v16h ah[4];
#pragma unroll
  for (int kc = 0; kc < 4; ++kc)
    ah[kc] = a_frag_h(hb + l15 * D_H + kc * 32 + khalf * 8);
#pragma unroll
  for (int nt = 0; nt < 4; ++nt) {
    v8f acc = vzero;
#pragma unroll
    for (int kc = 0; kc < 4; ++kc) {
      const half_t* bp = wt2 + (nt * 16 + l15) * D_H + kc * 32 + khalf * 8;
      v16h b = __builtin_shufflevector(*(const v8h*)(bp), *(const v8h*)(bp + 16), 0, 1, 2, 3, 4, 5, 6, 7, 8, 9, 10, 11, 12, 13, 14, 15);
      acc = wmma_f16(ah[kc], b, acc);
    }
    {
      v8h h;
#pragma unroll
      for (int i = 0; i < 8; ++i) h[i] = (half_t)acc[i];
      *(v8h*)(st + l15 * 16 + khalf * 8) = h;
    }
    v16h bs = hzero;
    bs.lo = *(const v8h*)(st + l15 * 16 + khalf * 8);
    v8f h2 = wmma_f16(afrag, bs, vzero);
    {
      int n = nt * 16 + l15;
      float bv = b2[n];
      int m0 = khalf * 8;
#pragma unroll
      for (int r = 0; r < 8; ++r) {
        float v = h2[r] + bv;
        hb[(m0 + r) * D_H2 + n] = (half_t)(v > 0.f ? v : 0.f);  // reuse hb as 16x64
      }
    }
  }

  // ============ Layer 3: adj @ (H2@W3) + b3, then mean over nodes ============
  v16h a3[2];
#pragma unroll
  for (int kc = 0; kc < 2; ++kc)
    a3[kc] = a_frag_h(hb + l15 * D_H2 + kc * 32 + khalf * 8);
#pragma unroll
  for (int nt = 0; nt < 2; ++nt) {
    v8f acc = vzero;
#pragma unroll
    for (int kc = 0; kc < 2; ++kc) {
      const half_t* bp = wt3 + (nt * 16 + l15) * D_H2 + kc * 32 + khalf * 8;
      v16h b = __builtin_shufflevector(*(const v8h*)(bp), *(const v8h*)(bp + 16), 0, 1, 2, 3, 4, 5, 6, 7, 8, 9, 10, 11, 12, 13, 14, 15);
      acc = wmma_f16(a3[kc], b, acc);
    }
    {
      v8h h;
#pragma unroll
      for (int i = 0; i < 8; ++i) h[i] = (half_t)acc[i];
      *(v8h*)(st + l15 * 16 + khalf * 8) = h;
    }
    v16h bs = hzero;
    bs.lo = *(const v8h*)(st + l15 * 16 + khalf * 8);
    v8f fin = wmma_f16(afrag, bs, vzero);
    // mean over the 16 nodes: sum this lane's 8 M-rows, add the other M-half
    float s = 0.f;
#pragma unroll
    for (int r = 0; r < 8; ++r) s += fin[r];
    // SWAPX16 ds_swizzle: combine lane l with lane l^16 (same column N)
    int sw = __builtin_amdgcn_ds_swizzle(__float_as_int(s), 0x401F);
    s += __int_as_float(sw);
    if (lane < 16) {
      int c = nt * 16 + lane;
      out[(size_t)g * D_OUT + c] = s * (1.0f / 16.0f) + b3[c];
    }
  }
}

extern "C" void kernel_launch(void* const* d_in, const int* in_sizes, int n_in,
                              void* d_out, int out_size, void* d_ws, size_t ws_size,
                              hipStream_t stream) {
  const float* X  = (const float*)d_in[0];
  const float* A  = (const float*)d_in[1];
  const float* W1 = (const float*)d_in[2];
  const float* b1 = (const float*)d_in[3];
  const float* W2 = (const float*)d_in[4];
  const float* b2 = (const float*)d_in[5];
  const float* W3 = (const float*)d_in[6];
  const float* b3 = (const float*)d_in[7];
  float* out = (float*)d_out;
  half_t* ws = (half_t*)d_ws;   // needs 86016 bytes

  gcn_prep<<<(D_H * D_IN + 255) / 256, 256, 0, stream>>>(W1, W2, W3, ws);
  gcn_main<<<B_GRAPHS / WAVES, 256, 0, stream>>>(X, A, b1, b2, b3, ws, out);
}
